// PointerNet_47210280517708
// MI455X (gfx1250) — compile-verified, matrix operand order checked by reading, not run on hardware
//
#include <hip/hip_runtime.h>
#include <hip/hip_bf16.h>
#include <cstdint>

// ---------------------------------------------------------------------------
// PointerNet on MI455X (gfx1250): all matmuls via v_wmma_f32_16x16x32_f16.
// B=256, L=128, E=512.
// ---------------------------------------------------------------------------

#define B_ 256
#define L_ 128
#define E_ 512
#define NEG_ (-1000000000.0f)

typedef __attribute__((ext_vector_type(16))) _Float16 v16h;
typedef __attribute__((ext_vector_type(8)))  float    v8f;

// Load one lane's 16-half WMMA operand slice from a row-major row.
// ISA 16-bit A/B layout: lanes 0-15 hold K = [kBase..kBase+7] and [16+kBase..16+kBase+7]
// with kBase=0; lanes 16-31 use kBase=8.
__device__ __forceinline__ v16h load_half_tile(const _Float16* __restrict__ row, int kBase) {
    v16h r;
    const _Float16* p0 = row + kBase;
    const _Float16* p1 = row + 16 + kBase;
#pragma unroll
    for (int j = 0; j < 8; ++j) { r[j] = p0[j]; r[j + 8] = p1[j]; }
    return r;
}

__device__ __forceinline__ float sigm(float x) { return 1.0f / (1.0f + __expf(-x)); }

// ---------------------------------------------------------------------------
// f32 -> f16 conversion
// ---------------------------------------------------------------------------
__global__ void k_cvt(const float* __restrict__ src, _Float16* __restrict__ dst, int n) {
    int i = blockIdx.x * blockDim.x + threadIdx.x;
    if (i < n) dst[i] = (_Float16)src[i];
}

// ---------------------------------------------------------------------------
// Embedding: emb[b,l,e] = in[b,l,0]*W[e,0] + in[b,l,1]*W[e,1] + bias[e]  (f16)
// Also initializes mask[b,l] = inputs[b,l,0] > 0.
// ---------------------------------------------------------------------------
__global__ void k_embed(const float* __restrict__ inp, const float* __restrict__ W,
                        const float* __restrict__ bias, _Float16* __restrict__ embH,
                        int* __restrict__ mask) {
    int idx = blockIdx.x * blockDim.x + threadIdx.x;
    if (idx >= B_ * L_ * E_) return;
    int e  = idx & (E_ - 1);
    int bl = idx >> 9;                 // E_ == 512
    float x0 = inp[(size_t)bl * 2 + 0];
    float x1 = inp[(size_t)bl * 2 + 1];
    float v  = x0 * W[e * 2 + 0] + x1 * W[e * 2 + 1] + bias[e];
    embH[idx] = (_Float16)v;
    if (e == 0) mask[bl] = (x0 > 0.0f) ? 1 : 0;
}

// ---------------------------------------------------------------------------
// State init: h = h0, c = c0 broadcast; h16 = f16(h0); x16 = f16(dec_input0)
// ---------------------------------------------------------------------------
__global__ void k_init(const float* __restrict__ h0, const float* __restrict__ c0,
                       const float* __restrict__ x0, float* __restrict__ h,
                       float* __restrict__ c, _Float16* __restrict__ h16,
                       _Float16* __restrict__ x16) {
    int idx = blockIdx.x * blockDim.x + threadIdx.x;
    if (idx >= B_ * E_) return;
    int e = idx & (E_ - 1);
    float hv = h0[e], cv = c0[e];
    h[idx] = hv; c[idx] = cv;
    h16[idx] = (_Float16)hv;
    x16[idx] = (_Float16)x0[e];
}

// ---------------------------------------------------------------------------
// One fused LSTM step. Block = 128 threads (4 waves). Block tile: 16 batch
// rows x 16 E cols. Wave g computes gate g via WMMA over K = E (x·Wih^T)
// + K = E (h·Whh^T), i.e. 32 x v_wmma_f32_16x16x32_f16 per wave. Gate tiles
// meet in LDS, then the block applies the LSTM nonlinearity + length mask.
// Grid: (B/16, E/16) = (16, 32).
// ---------------------------------------------------------------------------
__global__ __launch_bounds__(128)
void k_lstm_step(const _Float16* __restrict__ xbase, long xstride,
                 const _Float16* __restrict__ h16in,
                 const _Float16* __restrict__ Wih, const _Float16* __restrict__ Whh,
                 const float* __restrict__ bih, const float* __restrict__ bhh,
                 float* __restrict__ h, float* __restrict__ c,
                 _Float16* __restrict__ h16out,
                 const int* __restrict__ lengths, int t,
                 _Float16* __restrict__ enc_out) {
    __shared__ float gsh[4][16][16];

    const int bm    = blockIdx.x * 16;
    const int en    = blockIdx.y * 16;
    const int wave  = threadIdx.x >> 5;        // gate id: 0=i 1=f 2=g 3=o
    const int lane  = threadIdx.x & 31;
    const int mr    = lane & 15;
    const int kBase = (lane >> 4) * 8;
    const int gcol  = wave * E_ + en;          // gate column base in [4E,E] weights

    v8f acc = {0.f, 0.f, 0.f, 0.f, 0.f, 0.f, 0.f, 0.f};

    // x @ Wih^T  (A row = x[bm+mr, :], B row = Wih[gcol+mr, :])
    {
        const _Float16* arow = xbase + (size_t)(bm + mr) * xstride;
        const _Float16* brow = Wih + (size_t)(gcol + mr) * E_;
#pragma unroll 4
        for (int kc = 0; kc < E_; kc += 32) {
            __builtin_prefetch(arow + kc + 128, 0, 1);
            v16h a = load_half_tile(arow + kc, kBase);
            v16h b = load_half_tile(brow + kc, kBase);
            acc = __builtin_amdgcn_wmma_f32_16x16x32_f16(false, a, false, b,
                                                         (short)0, acc, false, false);
        }
    }
    // h @ Whh^T
    {
        const _Float16* arow = h16in + (size_t)(bm + mr) * E_;
        const _Float16* brow = Whh + (size_t)(gcol + mr) * E_;
#pragma unroll 4
        for (int kc = 0; kc < E_; kc += 32) {
            __builtin_prefetch(brow + kc + 128, 0, 1);
            v16h a = load_half_tile(arow + kc, kBase);
            v16h b = load_half_tile(brow + kc, kBase);
            acc = __builtin_amdgcn_wmma_f32_16x16x32_f16(false, a, false, b,
                                                         (short)0, acc, false, false);
        }
    }

    // Bias is per output column n = lane&15.
    const float bs    = bih[gcol + mr] + bhh[gcol + mr];
    const int   mbase = (lane >> 4) * 8;       // D layout: m = mbase + v, n = mr
#pragma unroll
    for (int v = 0; v < 8; ++v) gsh[wave][mbase + v][mr] = acc[v] + bs;
    __syncthreads();

    // Gate math: 256 tile elements over 128 threads.
    for (int eidx = threadIdx.x; eidx < 256; eidx += 128) {
        int m = eidx >> 4, n = eidx & 15;
        int bb = bm + m, ee = en + n;
        float gi = gsh[0][m][n], gf = gsh[1][m][n];
        float gg = gsh[2][m][n], go = gsh[3][m][n];
        size_t off  = (size_t)bb * E_ + ee;
        float cOld  = c[off];
        float hOld  = h[off];
        float cn    = sigm(gf) * cOld + sigm(gi) * tanhf(gg);
        float hn    = sigm(go) * tanhf(cn);
        bool valid  = true;
        if (lengths) valid = (t < lengths[bb]);
        float hN = valid ? hn : hOld;
        float cN = valid ? cn : cOld;
        h[off] = hN;
        c[off] = cN;
        h16out[off] = (_Float16)hN;
        if (enc_out)
            enc_out[((size_t)bb * L_ + t) * E_ + ee] = (_Float16)(valid ? hn : 0.0f);
    }
}

// ---------------------------------------------------------------------------
// Generic GEMM: out[m,n] = sum_k A[m,k] * Bw[n,k] + bias[n].
// Block = 128 threads (4 waves); wave w owns the 16x16 tile at column
// blockIdx.y*64 + w*16. Grid: (M/16, N/64).
// ---------------------------------------------------------------------------
__global__ __launch_bounds__(128)
void k_gemm_f16(const _Float16* __restrict__ A, long lda,
                const _Float16* __restrict__ Bw, long ldb,
                const float* __restrict__ bias,
                float* __restrict__ outF, _Float16* __restrict__ outH, long ldo,
                int K) {
    const int m0    = blockIdx.x * 16;
    const int wave  = threadIdx.x >> 5;
    const int n0    = blockIdx.y * 64 + wave * 16;
    const int lane  = threadIdx.x & 31;
    const int mr    = lane & 15;
    const int kBase = (lane >> 4) * 8;

    v8f acc = {0.f, 0.f, 0.f, 0.f, 0.f, 0.f, 0.f, 0.f};
    const _Float16* arow = A  + (size_t)(m0 + mr) * lda;
    const _Float16* brow = Bw + (size_t)(n0 + mr) * ldb;
#pragma unroll 4
    for (int kc = 0; kc < K; kc += 32) {
        __builtin_prefetch(arow + kc + 128, 0, 1);
        v16h a = load_half_tile(arow + kc, kBase);
        v16h b = load_half_tile(brow + kc, kBase);
        acc = __builtin_amdgcn_wmma_f32_16x16x32_f16(false, a, false, b,
                                                     (short)0, acc, false, false);
    }
    const float bs    = bias ? bias[n0 + mr] : 0.0f;
    const int   mbase = (lane >> 4) * 8;
#pragma unroll
    for (int v = 0; v < 8; ++v) {
        size_t off = (size_t)(m0 + mbase + v) * ldo + (n0 + mr);
        float  val = acc[v] + bs;
        if (outF) outF[off] = val;
        if (outH) outH[off] = (_Float16)val;
    }
}

// ---------------------------------------------------------------------------
// Attention scores: u[b,l] = sum_e tanh(W1enc[b,l,e] + q[b,e]) * V[e].
// Block = 256 threads (8 waves) per batch row b; q,V cached in LDS; one wave
// per l, 16 elements per lane, wave-level shuffle reduce.
// ---------------------------------------------------------------------------
__global__ __launch_bounds__(256)
void k_attn(const _Float16* __restrict__ W1enc, const float* __restrict__ q,
            const float* __restrict__ V, float* __restrict__ u) {
    __shared__ float qs[E_];
    __shared__ float vs[E_];
    const int b = blockIdx.x;
    for (int i = threadIdx.x; i < E_; i += 256) {
        qs[i] = q[(size_t)b * E_ + i];
        vs[i] = V[i];
    }
    __syncthreads();
    const int wave = threadIdx.x >> 5;
    const int lane = threadIdx.x & 31;
    for (int l = wave; l < L_; l += 8) {
        const _Float16* row = W1enc + ((size_t)b * L_ + l) * E_;
        float acc = 0.0f;
#pragma unroll
        for (int j = 0; j < 16; ++j) {
            int e = j * 32 + lane;               // coalesced, bank-conflict free
            acc += tanhf((float)row[e] + qs[e]) * vs[e];
        }
#pragma unroll
        for (int off = 16; off > 0; off >>= 1) acc += __shfl_xor(acc, off, 32);
        if (lane == 0) u[(size_t)b * L_ + l] = acc;
    }
}

// ---------------------------------------------------------------------------
// Masked softmax + argmax + mask update + next-input gather. One block of
// 128 threads per batch row. Writes probs[t,b,:] and idx[b,t] into d_out.
// ---------------------------------------------------------------------------
__global__ __launch_bounds__(128)
void k_select(const float* __restrict__ u, int* __restrict__ mask,
              const _Float16* __restrict__ embH, _Float16* __restrict__ x16,
              float* __restrict__ outProbs, float* __restrict__ outIdx, int t) {
    __shared__ float sv[128];
    __shared__ int   si[128];
    __shared__ float red[128];
    __shared__ int   sBest;
    __shared__ float sMax, sSum;

    const int b = blockIdx.x;
    const int l = threadIdx.x;
    const float uv = mask[b * L_ + l] ? u[(size_t)b * L_ + l] : NEG_;
    sv[l] = uv; si[l] = l;
    __syncthreads();
    // argmax with first-index tie break
    for (int s = 64; s > 0; s >>= 1) {
        if (l < s) {
            float v2 = sv[l + s]; int i2 = si[l + s];
            if (v2 > sv[l] || (v2 == sv[l] && i2 < si[l])) { sv[l] = v2; si[l] = i2; }
        }
        __syncthreads();
    }
    if (l == 0) { sBest = si[0]; sMax = sv[0]; }
    __syncthreads();
    const float ex = __expf(uv - sMax);
    red[l] = ex;
    __syncthreads();
    for (int s = 64; s > 0; s >>= 1) {
        if (l < s) red[l] += red[l + s];
        __syncthreads();
    }
    if (l == 0) sSum = red[0];
    __syncthreads();
    outProbs[(size_t)t * (B_ * L_) + (size_t)b * L_ + l] = ex / sSum;
    if (l == 0) {
        mask[b * L_ + sBest] = 0;
        outIdx[(size_t)b * L_ + t] = (float)sBest;
    }
    __syncthreads();
    const _Float16* src = embH + ((size_t)b * L_ + sBest) * E_;
    for (int j = l; j < E_; j += 128) x16[(size_t)b * E_ + j] = src[j];
}

// ---------------------------------------------------------------------------
// Host orchestration
// ---------------------------------------------------------------------------
extern "C" void kernel_launch(void* const* d_in, const int* in_sizes, int n_in,
                              void* d_out, int out_size, void* d_ws, size_t ws_size,
                              hipStream_t stream) {
    (void)in_sizes; (void)n_in; (void)out_size; (void)ws_size;

    const float* inputs  = (const float*)d_in[0];
    const int*   lengths = (const int*)d_in[1];
    const float* emb_W   = (const float*)d_in[2];
    const float* emb_b   = (const float*)d_in[3];
    const float* enc_Wih = (const float*)d_in[4];
    const float* enc_Whh = (const float*)d_in[5];
    const float* enc_bih = (const float*)d_in[6];
    const float* enc_bhh = (const float*)d_in[7];
    const float* dec_Wih = (const float*)d_in[8];
    const float* dec_Whh = (const float*)d_in[9];
    const float* dec_bih = (const float*)d_in[10];
    const float* dec_bhh = (const float*)d_in[11];
    const float* W1      = (const float*)d_in[12];
    const float* b1      = (const float*)d_in[13];
    const float* W2      = (const float*)d_in[14];
    const float* b2      = (const float*)d_in[15];
    const float* V       = (const float*)d_in[16];
    const float* h0      = (const float*)d_in[17];
    const float* c0      = (const float*)d_in[18];
    const float* x0      = (const float*)d_in[19];

    // --- workspace carve-up ---
    char* ws = (char*)d_ws;
    auto carve = [&](size_t bytes) -> char* {
        char* p = ws;
        ws += (bytes + 255) & ~(size_t)255;
        return p;
    };
    _Float16* embH    = (_Float16*)carve((size_t)B_ * L_ * E_ * 2);  // 32 MB
    _Float16* encH    = (_Float16*)carve((size_t)B_ * L_ * E_ * 2);  // 32 MB
    _Float16* W1encH  = (_Float16*)carve((size_t)B_ * L_ * E_ * 2);  // 32 MB
    _Float16* h16a    = (_Float16*)carve((size_t)B_ * E_ * 2);
    _Float16* h16b    = (_Float16*)carve((size_t)B_ * E_ * 2);
    _Float16* x16     = (_Float16*)carve((size_t)B_ * E_ * 2);
    float*    hbuf    = (float*)carve((size_t)B_ * E_ * 4);
    float*    cbuf    = (float*)carve((size_t)B_ * E_ * 4);
    float*    qbuf    = (float*)carve((size_t)B_ * E_ * 4);
    float*    ubuf    = (float*)carve((size_t)B_ * L_ * 4);
    int*      maskbuf = (int*)carve((size_t)B_ * L_ * 4);
    _Float16* encWihH = (_Float16*)carve((size_t)4 * E_ * E_ * 2);
    _Float16* encWhhH = (_Float16*)carve((size_t)4 * E_ * E_ * 2);
    _Float16* decWihH = (_Float16*)carve((size_t)4 * E_ * E_ * 2);
    _Float16* decWhhH = (_Float16*)carve((size_t)4 * E_ * E_ * 2);
    _Float16* W1H     = (_Float16*)carve((size_t)E_ * E_ * 2);
    _Float16* W2H     = (_Float16*)carve((size_t)E_ * E_ * 2);

    float* outProbs = (float*)d_out;
    float* outIdx   = (float*)d_out + (size_t)L_ * B_ * L_;

    // --- weight conversion (f32 -> f16) ---
    const int NW = 4 * E_ * E_;
    k_cvt<<<(NW + 255) / 256, 256, 0, stream>>>(enc_Wih, encWihH, NW);
    k_cvt<<<(NW + 255) / 256, 256, 0, stream>>>(enc_Whh, encWhhH, NW);
    k_cvt<<<(NW + 255) / 256, 256, 0, stream>>>(dec_Wih, decWihH, NW);
    k_cvt<<<(NW + 255) / 256, 256, 0, stream>>>(dec_Whh, decWhhH, NW);
    const int NS = E_ * E_;
    k_cvt<<<(NS + 255) / 256, 256, 0, stream>>>(W1, W1H, NS);
    k_cvt<<<(NS + 255) / 256, 256, 0, stream>>>(W2, W2H, NS);

    // --- embedding + mask + state init ---
    const int NE = B_ * L_ * E_;
    k_embed<<<(NE + 255) / 256, 256, 0, stream>>>(inputs, emb_W, emb_b, embH, maskbuf);
    const int NB = B_ * E_;
    k_init<<<(NB + 255) / 256, 256, 0, stream>>>(h0, c0, x0, hbuf, cbuf, h16a, x16);

    const dim3 glstm(B_ / 16, E_ / 16);   // (16, 32)
    const dim3 blstm(128);

    // --- encoder: 128 sequential fused LSTM steps ---
    for (int t = 0; t < L_; ++t) {
        _Float16* hin  = (t & 1) ? h16b : h16a;
        _Float16* hout = (t & 1) ? h16a : h16b;
        k_lstm_step<<<glstm, blstm, 0, stream>>>(
            embH + (size_t)t * E_, (long)L_ * E_, hin,
            encWihH, encWhhH, enc_bih, enc_bhh,
            hbuf, cbuf, hout, lengths, t, encH);
    }

    // --- W1enc = enc_out @ W1^T + b1  (stored f16) ---
    k_gemm_f16<<<dim3((B_ * L_) / 16, E_ / 64), 128, 0, stream>>>(
        encH, E_, W1H, E_, b1, nullptr, W1encH, E_, E_);

    // --- decoder: 128 sequential steps of LSTM -> q -> attention -> select ---
    for (int t = 0; t < L_; ++t) {
        _Float16* hin  = (t & 1) ? h16b : h16a;
        _Float16* hout = (t & 1) ? h16a : h16b;
        k_lstm_step<<<glstm, blstm, 0, stream>>>(
            x16, (long)E_, hin,
            decWihH, decWhhH, dec_bih, dec_bhh,
            hbuf, cbuf, hout, nullptr, t, nullptr);
        k_gemm_f16<<<dim3(B_ / 16, E_ / 64), 128, 0, stream>>>(
            hout, E_, W2H, E_, b2, qbuf, nullptr, E_, E_);
        k_attn<<<B_, 256, 0, stream>>>(W1encH, qbuf, V, ubuf);
        k_select<<<B_, 128, 0, stream>>>(ubuf, maskbuf, embH, x16,
                                         outProbs, outIdx, t);
    }
}
